// MoEBertLayer_38534446579848
// MI455X (gfx1250) — matrix operand order repaired and checked by reading, not compiled, not run on hardware
//
#include <hip/hip_runtime.h>
#include <math.h>

// ---------------------------------------------------------------------------
// MoE-BERT layer forward for MI455X (gfx1250), wave32, bf16 WMMA w/ f32 acc.
//
// Roofline: 64 GFLOP total, 113MB of f32 weights (fits 192MB L2 -> one HBM
// pass ~5us @ 23.3TB/s). Compute-bound -> v_wmma_f32_16x16x32_bf16 everywhere,
// f32->bf16 conversion fused into LDS staging. GEMM blocks use 128x64 tiles
// with 2x2 subtiles/wave so each A/B fragment feeds two WMMAs.
// ---------------------------------------------------------------------------

typedef __attribute__((ext_vector_type(16))) __bf16 v16bf;
typedef __attribute__((ext_vector_type(8)))  float  v8f;

#define NB  8      // batch
#define SS  512    // seq len
#define HH  768    // hidden
#define NHD 12     // heads
#define DHD 64     // head dim
#define FFD 3072   // intermediate

__device__ __forceinline__ float gelu_exact(float x) {
    return 0.5f * x * (1.0f + erff(x * 0.70710678118654752440f));
}

// ---------------------------------------------------------------------------
// Generic per-sample expert GEMM:  Y[b] = act( X[b] * W[e_b] + bias[e_b] )
//   X: [B, M, K] f32 row-major, W: [E, K, N] f32 row-major, bias: [E, N]
// Block = 256 threads (8 waves), 128x64 tile, K-step 32.
// Each wave -> 2x2 16x16 WMMA subtiles (A,B fragments reused twice each).
// Fragment layouts per CDNA5 ISA 7.12.2:
//   A (16-bit 16x32): lane = m + 16*half; a[2v+s] -> k = 16*(v/4)+8*half+2*(v%4)+s
//   B (32x16): lane = n + 16*half;         b[2v+s] -> k = 16*(v/4)+8*half+2*(v%4)+s
//   C/D f32 16x16: elem r -> m = r + 8*half, n = lane%16
// ---------------------------------------------------------------------------
template<int ACT>  // 0 = none, 1 = exact GELU
__global__ __launch_bounds__(256) void expert_gemm_kernel(
    const float* __restrict__ X, const float* __restrict__ W,
    const float* __restrict__ bias, const int* __restrict__ eidx,
    float* __restrict__ Y, int M, int K, int N)
{
    __shared__ __bf16 As[128][32];  // [m][k]                       (8 KB)
    __shared__ __bf16 Bs[64][32];   // [n][k] transposed staging    (4 KB)

    const int b = blockIdx.z;
    const int e = eidx[b];
    const float* Xb = X + (size_t)b * M * K;
    const float* We = W + (size_t)e * K * N;
    const float* be = bias + (size_t)e * N;
    float* Yb = Y + (size_t)b * M * N;

    const int m0 = blockIdx.y * 128;
    const int n0 = blockIdx.x * 64;
    const int tid = threadIdx.x;
    const int wave = tid >> 5, lane = tid & 31;
    const int lh = lane >> 4, lr = lane & 15;
    const int smr = (wave >> 1) * 2;      // m-subtile base: 0,2,4,6
    const int snc = (wave & 1) * 2;       // n-subtile base: 0,2

    v8f acc00 = {}, acc01 = {}, acc10 = {}, acc11 = {};

    for (int k0 = 0; k0 < K; k0 += 32) {
        // --- stage A tile 128x32 (4096 f32 = 1024 float4, 4 per thread) ---
        #pragma unroll
        for (int j = 0; j < 4; ++j) {
            int cid = tid + 256 * j;
            int r = cid >> 3, cc = cid & 7;
            const float* src = Xb + (size_t)(m0 + r) * K + k0 + cc * 4;
            if (k0 + 32 < K) __builtin_prefetch(src + 32, 0, 1);
            float4 f = *(const float4*)src;
            __bf16* dst = &As[r][cc * 4];
            dst[0] = (__bf16)f.x; dst[1] = (__bf16)f.y;
            dst[2] = (__bf16)f.z; dst[3] = (__bf16)f.w;
        }
        // --- stage B tile 32x64, transposed into Bs[n][k] ---
        #pragma unroll
        for (int j = 0; j < 2; ++j) {
            int cid = tid + 256 * j;
            int r = cid >> 4, cc = cid & 15;   // r = k row (0..31), cc = n chunk
            const float* src = We + (size_t)(k0 + r) * N + n0 + cc * 4;
            if (k0 + 32 < K) __builtin_prefetch(src + (size_t)32 * N, 0, 1);
            float4 f = *(const float4*)src;
            Bs[cc*4+0][r] = (__bf16)f.x; Bs[cc*4+1][r] = (__bf16)f.y;
            Bs[cc*4+2][r] = (__bf16)f.z; Bs[cc*4+3][r] = (__bf16)f.w;
        }
        __syncthreads();

        v16bf a0, a1, b0, b1;
        #pragma unroll
        for (int v = 0; v < 8; ++v) {
            int kk = 16*(v>>2) + 8*lh + 2*(v&3);
            a0[2*v] = As[smr*16 + lr][kk];      a0[2*v+1] = As[smr*16 + lr][kk+1];
            a1[2*v] = As[(smr+1)*16 + lr][kk];  a1[2*v+1] = As[(smr+1)*16 + lr][kk+1];
        }
        #pragma unroll
        for (int v = 0; v < 8; ++v) {
            int kk = 16*(v>>2) + 8*lh + 2*(v&3);
            b0[2*v] = Bs[snc*16 + lr][kk];      b0[2*v+1] = Bs[snc*16 + lr][kk+1];
            b1[2*v] = Bs[(snc+1)*16 + lr][kk];  b1[2*v+1] = Bs[(snc+1)*16 + lr][kk+1];
        }
        acc00 = __builtin_amdgcn_wmma_f32_16x16x32_bf16(false, a0, false, b0, (short)0, acc00, false, false);
        acc01 = __builtin_amdgcn_wmma_f32_16x16x32_bf16(false, a0, false, b1, (short)0, acc01, false, false);
        acc10 = __builtin_amdgcn_wmma_f32_16x16x32_bf16(false, a1, false, b0, (short)0, acc10, false, false);
        acc11 = __builtin_amdgcn_wmma_f32_16x16x32_bf16(false, a1, false, b1, (short)0, acc11, false, false);
        __syncthreads();
    }

    // --- epilogue: bias (+GELU), scatter f32 stores ---
    #pragma unroll
    for (int i = 0; i < 2; ++i) {
        #pragma unroll
        for (int t = 0; t < 2; ++t) {
            v8f acc = (i == 0) ? (t == 0 ? acc00 : acc01)
                               : (t == 0 ? acc10 : acc11);
            int n = n0 + (snc + t) * 16 + lr;
            float bvv = be[n];
            #pragma unroll
            for (int r = 0; r < 8; ++r) {
                int m = m0 + (smr + i) * 16 + r + 8 * lh;
                float x = acc[r] + bvv;
                if (ACT == 1) x = gelu_exact(x);
                Yb[(size_t)m * N + n] = x;
            }
        }
    }
}

// ---------------------------------------------------------------------------
// Fused attention: per (b, head, 64 query rows). Flash-style online softmax,
// 32-key chunks. scores = (q.k^T)/8 + mask; ctx accumulated with row rescale.
// Static LDS ~29KB, all GEMM work through WMMA.
// ---------------------------------------------------------------------------
__global__ __launch_bounds__(256) void attn_fused_kernel(
    const float* __restrict__ Q, const float* __restrict__ Kt,
    const float* __restrict__ V, const float* __restrict__ mask,
    float* __restrict__ O)
{
    __shared__ __bf16 qs[64][64];    // [q_row][d]
    __shared__ __bf16 ks[32][64];    // [key][d]        (B-frag layout for scores)
    __shared__ __bf16 vs[64][32];    // [d][key]        (B-frag layout for ctx)
    __shared__ float  sc[64][32];    // raw scores this chunk
    __shared__ __bf16 ps[64][32];    // exp()'d probs (unnormalized)
    __shared__ float  row_max[64], row_sum[64], row_scale[64];

    const int bh = blockIdx.y;
    const int b = bh / NHD, h = bh % NHD;
    const int q0 = blockIdx.x * 64;
    const int tid = threadIdx.x;
    const int wave = tid >> 5, lane = tid & 31;
    const int lh = lane >> 4, lr = lane & 15;
    const size_t base = ((size_t)b * SS) * HH + (size_t)h * DHD;

    // stage Q tile 64x64 (4096 f32 = 1024 float4, 4 per thread)
    #pragma unroll
    for (int j = 0; j < 4; ++j) {
        int cid = tid + 256 * j;
        int r = cid >> 4, cc = cid & 15;
        float4 f = *(const float4*)(Q + base + (size_t)(q0 + r) * HH + cc * 4);
        qs[r][cc*4+0] = (__bf16)f.x; qs[r][cc*4+1] = (__bf16)f.y;
        qs[r][cc*4+2] = (__bf16)f.z; qs[r][cc*4+3] = (__bf16)f.w;
    }
    if (tid < 64) { row_max[tid] = -3.0e38f; row_sum[tid] = 0.0f; }

    v8f acc0 = {}, acc1 = {};
    const int sm_s = wave >> 1;         // scores: m subtile 0..3
    const int sn_s = wave & 1;          // scores: n subtile 0..1 (32 keys)
    const int cm = wave >> 1;           // ctx: m subtile row
    const int cn0 = (wave & 1) * 2;     // ctx: n subtile pair

    for (int kb = 0; kb < SS / 32; ++kb) {
        __syncthreads();
        // stage K chunk (32 keys x 64 d) -> ks[key][d]
        #pragma unroll
        for (int j = 0; j < 2; ++j) {
            int cid = tid + 256 * j;
            int r = cid >> 4, cc = cid & 15;
            float4 f = *(const float4*)(Kt + base + (size_t)(kb*32 + r) * HH + cc * 4);
            ks[r][cc*4+0] = (__bf16)f.x; ks[r][cc*4+1] = (__bf16)f.y;
            ks[r][cc*4+2] = (__bf16)f.z; ks[r][cc*4+3] = (__bf16)f.w;
        }
        // stage V chunk transposed -> vs[d][key]
        #pragma unroll
        for (int j = 0; j < 2; ++j) {
            int cid = tid + 256 * j;
            int r = cid >> 4, cc = cid & 15;
            float4 f = *(const float4*)(V + base + (size_t)(kb*32 + r) * HH + cc * 4);
            vs[cc*4+0][r] = (__bf16)f.x; vs[cc*4+1][r] = (__bf16)f.y;
            vs[cc*4+2][r] = (__bf16)f.z; vs[cc*4+3][r] = (__bf16)f.w;
        }
        __syncthreads();

        // scores 16x16 subtile, K = DHD = 64 -> two 32-deep WMMAs
        v8f s = {};
        #pragma unroll
        for (int ks2 = 0; ks2 < 64; ks2 += 32) {
            v16bf a, bb;
            int am = sm_s * 16 + lr;
            #pragma unroll
            for (int v = 0; v < 8; ++v) {
                int kk = ks2 + 16*(v>>2) + 8*lh + 2*(v&3);
                a[2*v] = qs[am][kk]; a[2*v+1] = qs[am][kk+1];
            }
            int bn = sn_s * 16 + lr;
            #pragma unroll
            for (int v = 0; v < 8; ++v) {
                int kk = ks2 + 16*(v>>2) + 8*lh + 2*(v&3);
                bb[2*v] = ks[bn][kk]; bb[2*v+1] = ks[bn][kk+1];
            }
            s = __builtin_amdgcn_wmma_f32_16x16x32_bf16(
                    false, a, false, bb, (short)0, s, false, false);
        }
        // scaled + masked scores -> LDS
        {
            int col = sn_s * 16 + lr;
            float mval = mask[(size_t)b * SS + kb*32 + col];
            #pragma unroll
            for (int r = 0; r < 8; ++r)
                sc[sm_s*16 + r + 8*lh][col] = s[r] * 0.125f + mval;
        }
        __syncthreads();

        // online softmax update, one thread per query row
        if (tid < 64) {
            float mxo = row_max[tid], nm = mxo;
            #pragma unroll 4
            for (int j = 0; j < 32; ++j) nm = fmaxf(nm, sc[tid][j]);
            float f = __expf(mxo - nm);          // first chunk: exp(-huge) = 0
            float sum = row_sum[tid] * f;
            for (int j = 0; j < 32; ++j) {
                float p = __expf(sc[tid][j] - nm);
                ps[tid][j] = (__bf16)p;
                sum += p;
            }
            row_max[tid] = nm; row_sum[tid] = sum; row_scale[tid] = f;
        }
        __syncthreads();

        // rescale running ctx accumulators
        #pragma unroll
        for (int r = 0; r < 8; ++r) {
            float f = row_scale[cm*16 + r + 8*lh];
            acc0[r] *= f; acc1[r] *= f;
        }
        // ctx += P(16x32) x V(32x16,16) : one K-step of 32 keys
        {
            v16bf a;
            int am = cm * 16 + lr;
            #pragma unroll
            for (int v = 0; v < 8; ++v) {
                int kk = 16*(v>>2) + 8*lh + 2*(v&3);
                a[2*v] = ps[am][kk]; a[2*v+1] = ps[am][kk+1];
            }
            #pragma unroll
            for (int t = 0; t < 2; ++t) {
                v16bf bb;
                int bn = (cn0 + t) * 16 + lr;
                #pragma unroll
                for (int v = 0; v < 8; ++v) {
                    int kk = 16*(v>>2) + 8*lh + 2*(v&3);
                    bb[2*v] = vs[bn][kk]; bb[2*v+1] = vs[bn][kk+1];
                }
                if (t == 0) acc0 = __builtin_amdgcn_wmma_f32_16x16x32_bf16(
                                       false, a, false, bb, (short)0, acc0, false, false);
                else        acc1 = __builtin_amdgcn_wmma_f32_16x16x32_bf16(
                                       false, a, false, bb, (short)0, acc1, false, false);
            }
        }
    }
    __syncthreads();

    // final 1/sum normalize, write ctx in [B,S,H] layout (head-interleaved)
    #pragma unroll
    for (int t = 0; t < 2; ++t) {
        v8f acc = t ? acc1 : acc0;
        int n = (cn0 + t) * 16 + lr;
        #pragma unroll
        for (int r = 0; r < 8; ++r) {
            int m = cm * 16 + r + 8*lh;
            O[base + (size_t)(q0 + m) * HH + n] = acc[r] * (1.0f / row_sum[m]);
        }
    }
}

// ---------------------------------------------------------------------------
// Y = LayerNorm(X + R) * g + beta, one block per row of 768 (= 3*256)
// ---------------------------------------------------------------------------
__global__ __launch_bounds__(256) void add_ln_kernel(
    const float* __restrict__ X, const float* __restrict__ R,
    const float* __restrict__ g, const float* __restrict__ bta,
    float* __restrict__ Y)
{
    __shared__ float red[256];
    const int row = blockIdx.x;
    const int tid = threadIdx.x;
    const size_t off = (size_t)row * HH;

    float v0 = X[off + tid]       + R[off + tid];
    float v1 = X[off + tid + 256] + R[off + tid + 256];
    float v2 = X[off + tid + 512] + R[off + tid + 512];

    red[tid] = v0 + v1 + v2;  __syncthreads();
    for (int o = 128; o > 0; o >>= 1) {
        if (tid < o) red[tid] += red[tid + o];
        __syncthreads();
    }
    float mu = red[0] * (1.0f / HH);
    __syncthreads();
    float d0 = v0 - mu, d1 = v1 - mu, d2 = v2 - mu;
    red[tid] = d0*d0 + d1*d1 + d2*d2;  __syncthreads();
    for (int o = 128; o > 0; o >>= 1) {
        if (tid < o) red[tid] += red[tid + o];
        __syncthreads();
    }
    float inv = rsqrtf(red[0] * (1.0f / HH) + 1e-12f);
    Y[off + tid]       = d0 * inv * g[tid]       + bta[tid];
    Y[off + tid + 256] = d1 * inv * g[tid + 256] + bta[tid + 256];
    Y[off + tid + 512] = d2 * inv * g[tid + 512] + bta[tid + 512];
}

// ---------------------------------------------------------------------------
extern "C" void kernel_launch(void* const* d_in, const int* in_sizes, int n_in,
                              void* d_out, int out_size, void* d_ws, size_t ws_size,
                              hipStream_t stream) {
    (void)in_sizes; (void)n_in; (void)out_size; (void)ws_size;
    const float* hs   = (const float*)d_in[0];
    const int*   eidx = (const int*)  d_in[1];
    const float* amsk = (const float*)d_in[2];
    const float* Wq = (const float*)d_in[3];  const float* bq = (const float*)d_in[4];
    const float* Wk = (const float*)d_in[5];  const float* bk = (const float*)d_in[6];
    const float* Wv = (const float*)d_in[7];  const float* bv = (const float*)d_in[8];
    const float* Wo = (const float*)d_in[9];  const float* bo = (const float*)d_in[10];
    const float* g1 = (const float*)d_in[11]; const float* b1 = (const float*)d_in[12];
    const float* Wi = (const float*)d_in[13]; const float* bi = (const float*)d_in[14];
    const float* Wt = (const float*)d_in[15]; const float* bt = (const float*)d_in[16];
    const float* g2 = (const float*)d_in[17]; const float* b2 = (const float*)d_in[18];
    float* out = (float*)d_out;

    // workspace layout (needs 4*12.58MB + 50.3MB ~= 96MB)
    char* ws = (char*)d_ws;
    const size_t ABYTES = (size_t)NB * SS * HH * sizeof(float);  // 12,582,912
    float* q    = (float*)(ws + 0 * ABYTES);
    float* kbuf = (float*)(ws + 1 * ABYTES);
    float* vbuf = (float*)(ws + 2 * ABYTES);
    float* ctx  = (float*)(ws + 3 * ABYTES);
    float* hmid = (float*)(ws + 4 * ABYTES);                     // B*S*FF f32
    float* attn = q;     // q dead after attention
    float* x1   = kbuf;  // k dead after attention
    float* ffn  = vbuf;  // v dead after attention

    dim3 blk(256);
    dim3 gHH(HH / 64, SS / 128, NB);

    expert_gemm_kernel<0><<<gHH, blk, 0, stream>>>(hs, Wq, bq, eidx, q,    SS, HH, HH);
    expert_gemm_kernel<0><<<gHH, blk, 0, stream>>>(hs, Wk, bk, eidx, kbuf, SS, HH, HH);
    expert_gemm_kernel<0><<<gHH, blk, 0, stream>>>(hs, Wv, bv, eidx, vbuf, SS, HH, HH);

    dim3 gA(SS / 64, NB * NHD);
    attn_fused_kernel<<<gA, blk, 0, stream>>>(q, kbuf, vbuf, amsk, ctx);

    expert_gemm_kernel<0><<<gHH, blk, 0, stream>>>(ctx, Wo, bo, eidx, attn, SS, HH, HH);
    add_ln_kernel<<<dim3(NB * SS), blk, 0, stream>>>(attn, hs, g1, b1, x1);

    dim3 gFF(FFD / 64, SS / 128, NB);
    expert_gemm_kernel<1><<<gFF, blk, 0, stream>>>(x1, Wi, bi, eidx, hmid, SS, HH, FFD);
    expert_gemm_kernel<0><<<gHH, blk, 0, stream>>>(hmid, Wt, bt, eidx, ffn, SS, FFD, HH);
    add_ln_kernel<<<dim3(NB * SS), blk, 0, stream>>>(ffn, x1, g2, b2, out);
}
